// GBM_88210038326292
// MI455X (gfx1250) — matrix operand order repaired and checked by reading, not compiled, not run on hardware
//
#include <hip/hip_runtime.h>

typedef __attribute__((ext_vector_type(16))) _Float16 v16h;
typedef __attribute__((ext_vector_type(8)))  float    v8f;

#define TT    500
#define KWIN  3
#define CCH   3
#define HH    128
#define WWI   128
#define HWP   (HH * WWI)          /* 16384 */
#define CHW   (CCH * HH * WWI)    /* 49152 */
#define NELEM (2 * CCH * HH * WWI)/* 98304 */
#define EPSV  1e-7f

// One block = (batch, image row). 8 waves x 16 pixels = 128 pixels (full row).
// Each wave: im2col 16x32 (f16) x Wm^T 32x16 (f16) -> v_wmma_f32_16x16x32_f16.
__global__ __launch_bounds__(256)
void gbm_fused(const float* __restrict__ x, const float* __restrict__ z,
               const float* __restrict__ Wm, const float* __restrict__ bm,
               const float* __restrict__ temb_w, const int* __restrict__ t,
               float* __restrict__ partials)
{
    __shared__ float s_xin[3 * 3 * 130];   // [ci][r in h-1..h+1][col+1], halo-padded
    __shared__ float s_err[8][16][6];      // per-wave conv outputs (6 real channels)
    __shared__ float s_red[8];

    const int tid  = threadIdx.x;
    const int wave = tid >> 5;
    const int lane = tid & 31;
    const int bidx = blockIdx.x;
    const int b = bidx >> 7;               // batch
    const int h = bidx & 127;              // image row

    const int tb = t[b];
    const float* xb    = x  + (size_t)b * TT * CHW;
    const float* xin_g = xb + (size_t)(tb - 1) * CHW;   // x_in = window[K-1]

    // ---- stage x_in rows h-1..h+1 with column halo into LDS (zero-padded SAME) ----
    for (int e = tid; e < 3 * 3 * 130; e += 256) {
        int ci  = e / 390;
        int rem = e - ci * 390;
        int r   = rem / 130;
        int col = rem - r * 130 - 1;       // -1 .. 128
        int srow = h - 1 + r;
        float v = 0.0f;
        if (srow >= 0 && srow < HH && col >= 0 && col < WWI)
            v = xin_g[ci * HWP + srow * WWI + col];
        s_xin[e] = v;
    }
    __syncthreads();

    // ---- build A (im2col, 16 pixels x K=32) and B (Wm^T, K=32 x 16) per ISA layout ----
    const int m    = lane & 15;
    const int hi   = lane >> 4;            // lane half selects K groups
    const int wcol = wave * 16 + m;        // pixel column for A row m

    v16h amat, bmat;
    #pragma unroll
    for (int hh = 0; hh < 16; ++hh) {
        // A: halves 0..7 -> K = hh (+8 if upper lanes); halves 8..15 -> K = hh+8 (+8)
        int ka = hh + ((hh >= 8) ? 8 : 0) + (hi ? 8 : 0);
        float av = 0.0f;
        if (ka < 27) {
            int ci  = ka / 9;
            int rr  = (ka - ci * 9) / 3;
            int cc2 = ka - ci * 9 - rr * 3;
            av = s_xin[(ci * 3 + rr) * 130 + wcol + cc2];   // halo makes this branch-free
        }
        amat[hh] = (_Float16)av;

        // B: N = lane&15, halves 0..7 -> K = hh (+8 if upper lanes); halves 8..15 -> K = hh+8 (+8)
        int kb = hh + ((hh >= 8) ? 8 : 0) + (hi ? 8 : 0);
        float bv = 0.0f;
        if (m < 6 && kb < 27) bv = Wm[m * 27 + kb];         // B[k][n] = Wm[n][k]
        bmat[hh] = (_Float16)bv;
    }

    v8f acc = {};
    acc = __builtin_amdgcn_wmma_f32_16x16x32_f16(
        /*neg_a=*/false, amat, /*neg_b=*/false, bmat,
        /*c_mod=*/(short)0, acc, /*reuse_a=*/false, /*reuse_b=*/false);

    // ---- err = conv + bias + temb, staged to LDS for the 6 real channels ----
    const float tscale = (float)tb / (float)TT;
    float badd = 0.0f, tadd = 0.0f;
    if (m < 6) { badd = bm[m]; tadd = temb_w[m] * tscale; }
    #pragma unroll
    for (int r = 0; r < 8; ++r) {
        int mrow = r + (hi ? 8 : 0);       // C/D layout: VGPR r, upper lanes = M+8
        if (m < 6) s_err[wave][mrow][m] = acc[r] + badd + tadd;
    }
    __syncthreads();

    // ---- pointwise: 48 tasks per wave = 16 pixels x 3 channels ----
    float lsum = 0.0f;
    #pragma unroll
    for (int pass = 0; pass < 2; ++pass) {
        int task = lane + pass * 32;
        if (task < 48) {
            int mm = task & 15;
            int c  = task >> 4;
            int wpix = wave * 16 + mm;
            int idx  = c * HWP + h * WWI + wpix;

            const float* wp = xb + (size_t)(tb - KWIN) * CHW + idx;
            float v0 = wp[0 * CHW], v1 = wp[1 * CHW], v2 = wp[2 * CHW],
                  v3 = wp[3 * CHW], v4 = wp[4 * CHW], v5 = wp[5 * CHW],
                  v6 = wp[6 * CHW];

            float mean = (v0 + v1 + v2 + v3 + v4 + v5 + v6) * (1.0f / 7.0f);
            float d0 = v0 - mean, d1 = v1 - mean, d2 = v2 - mean, d3 = v3 - mean,
                  d4 = v4 - mean, d5 = v5 - mean, d6 = v6 - mean;
            float var = (d0*d0 + d1*d1 + d2*d2 + d3*d3 + d4*d4 + d5*d5 + d6*d6)
                        * (1.0f / 6.0f);                       // ddof = 1
            float gs = fmaxf(sqrtf(var), EPSV);                // clip(std, EPS)

            float xin_v  = v2;                                  // window[K-1]
            float target = v3;                                  // window[K]
            float pmu = s_err[wave][mm][c];
            float ps  = fmaxf(s_err[wave][mm][c + 3], EPSV);    // clip(p_sigma, EPS)

            float wt = sqrtf((float)tb) * z[(size_t)b * CHW + idx];
            float xt = xin_v * expf(pmu + 0.5f * ps * ps + ps * wt);
            float l1 = fabsf(xt - target);

            float vr  = ps / gs;  vr  = vr * vr;
            float dmu = (pmu - mean) / gs;
            float kl  = 0.5f * (vr + dmu * dmu - 1.0f - logf(vr));

            lsum += l1 + kl;     // identical denominators -> single accumulator
        }
    }

    // ---- deterministic reduction: wave shfl -> LDS -> block partial ----
    for (int off = 16; off > 0; off >>= 1)
        lsum += __shfl_down(lsum, off, 32);
    if (lane == 0) s_red[wave] = lsum;
    __syncthreads();
    if (tid == 0) {
        float s = 0.0f;
        #pragma unroll
        for (int i = 0; i < 8; ++i) s += s_red[i];
        partials[bidx] = s;
    }
}

__global__ __launch_bounds__(256)
void gbm_finalize(const float* __restrict__ partials, float* __restrict__ out)
{
    __shared__ float s_red[8];
    int tid = threadIdx.x;
    float s = partials[tid];               // exactly 256 partials
    for (int off = 16; off > 0; off >>= 1)
        s += __shfl_down(s, off, 32);
    if ((tid & 31) == 0) s_red[tid >> 5] = s;
    __syncthreads();
    if (tid == 0) {
        float tot = 0.0f;
        #pragma unroll
        for (int i = 0; i < 8; ++i) tot += s_red[i];
        out[0] = tot * (1.0f / (float)NELEM);   // mean(l1) + mean(kl), same N
    }
}

extern "C" void kernel_launch(void* const* d_in, const int* in_sizes, int n_in,
                              void* d_out, int out_size, void* d_ws, size_t ws_size,
                              hipStream_t stream)
{
    const float* x      = (const float*)d_in[0];
    const float* z      = (const float*)d_in[1];
    const float* Wm     = (const float*)d_in[2];
    const float* bm     = (const float*)d_in[3];
    const float* temb_w = (const float*)d_in[4];
    const int*   t      = (const int*)d_in[5];
    float* partials = (float*)d_ws;        // 256 floats, fully overwritten each call

    gbm_fused<<<256, 256, 0, stream>>>(x, z, Wm, bm, temb_w, t, partials);
    gbm_finalize<<<1, 256, 0, stream>>>(partials, (float*)d_out);
}
